// DCN_9474697855729
// MI455X (gfx1250) — matrix operand order repaired and checked by reading, not compiled, not run on hardware
//
#include <hip/hip_runtime.h>

// ---------------------------------------------------------------------------
// CDNA5 (gfx1250) deep-clustering-network forward.
//   - weights pre-transposed+converted to bf16 [N][K]; activations kept bf16
//   - GEMM: TDM (tensor_load_to_lds) double-buffered staging into padded LDS,
//     v_wmma_f32_16x16x32_bf16 compute, f32 accum, bias+ReLU epilogue
//   - S = E*centers^T via same GEMM, then per-row argmin -> one-hot labels
// ---------------------------------------------------------------------------

typedef __attribute__((ext_vector_type(16))) __bf16 v16bf;
typedef __attribute__((ext_vector_type(8)))  float  v8f;
typedef __attribute__((ext_vector_type(4)))  unsigned int u32x4;
typedef __attribute__((ext_vector_type(8)))  unsigned int u32x8;

union Frag16 { v16bf v; uint4 q[2]; };

__device__ __forceinline__ unsigned int pack2bf(float a, float b) {
  union { __bf16 h; unsigned short u; } ca, cb;
  ca.h = (__bf16)a; cb.h = (__bf16)b;
  return (unsigned int)ca.u | ((unsigned int)cb.u << 16);
}
__device__ __forceinline__ unsigned short f2bf(float a) {
  union { __bf16 h; unsigned short u; } c;
  c.h = (__bf16)a;
  return c.u;
}

#define LDS_STRIDE 40          // 32 bf16 payload + 8 bf16 pad -> 80B rows
#define TILE_BYTES (128 * LDS_STRIDE * 2)   // 10240 B per tile buffer

// Issue one TDM 2D tile load: 128 rows x 32 bf16 from row-major [Nrows][Kd]
// global tensor (tile start pre-offset into gtile) into LDS at ldsByteOff,
// with 16B padding after every 64B row (matches LDS_STRIDE).
__device__ __forceinline__ void tdm_issue(const unsigned short* gtile,
                                          unsigned ldsByteOff, int Kd, int Nrows) {
  unsigned long long ga = (unsigned long long)(const void*)gtile;
  u32x4 g0;
  g0[0] = 1u;                                    // count=1, user descriptor
  g0[1] = ldsByteOff;                            // lds_addr (bytes)
  g0[2] = (unsigned)ga;                          // global_addr[31:0]
  g0[3] = ((unsigned)(ga >> 32) & 0x01FFFFFFu)   // global_addr[56:32]
          | (2u << 30);                          // type = 2 ("image")
  u32x8 g1;
  g1[0] = (1u << 16)     // data_size = 2 bytes
        | (1u << 20)     // pad_enable
        | (3u << 22)     // pad_interval: 2^(3+1)=16 DWORDs (= one 64B row)
        | (3u << 25);    // pad_amount: 3+1 = 4 DWORDs (16B)
  g1[1] = ((unsigned)Kd & 0xFFFFu) << 16;                      // tensor_dim0[15:0]
  g1[2] = ((unsigned)Kd >> 16) | (((unsigned)Nrows & 0xFFFFu) << 16); // td0 hi, td1 lo
  g1[3] = ((unsigned)Nrows >> 16) | (32u << 16);               // td1 hi, tile_dim0=32
  g1[4] = 128u;                                                // tile_dim1=128, tile_dim2=0
  g1[5] = (unsigned)Kd;                                        // tensor_dim0_stride lo
  g1[6] = 0u;                                                  // stride0 hi, stride1 lo
  g1[7] = 0u;
  asm volatile("tensor_load_to_lds %0, %1" :: "s"(g0), "s"(g1) : "memory");
}

// C = act(A * B^T + bias): A bf16 [M][Kd], B bf16 [N][Kd] (both k-contiguous)
template<int RELU, int BIAS, int WF32, int WBF16>
__global__ __launch_bounds__(256)
void gemm_tdm(const unsigned short* __restrict__ A,
              const unsigned short* __restrict__ B,
              const float* __restrict__ bias,
              float* __restrict__ Cf,
              unsigned short* __restrict__ Cbf,
              int M, int N, int Kd)
{
  extern __shared__ unsigned short smem[];
  // layout (byte offsets): As[0]=0, As[1]=10240, Bs[0]=20480, Bs[1]=30720

  const int tid  = threadIdx.x;
  const int lane = tid & 31;
  const int wid  = tid >> 5;
  const int lr   = lane & 15;
  const int hi   = lane >> 4;

  const int blockN = blockIdx.x * 128;
  const int blockM = blockIdx.y * 128;

  const int waveRow = (wid & 1) * 64;   // 2 waves x 64 rows
  const int waveCol = (wid >> 1) * 32;  // 4 waves x 32 cols

  v8f acc[4][2];
  #pragma unroll
  for (int mi = 0; mi < 4; ++mi)
    #pragma unroll
    for (int ni = 0; ni < 2; ++ni) {
      v8f z{};
      acc[mi][ni] = z;
    }

  const int nK = Kd >> 5;

  // prologue: stage tile 0 into buffer 0
  if (wid == 0) {
    tdm_issue(A + (size_t)blockM * Kd, 0u,              Kd, M);
    tdm_issue(B + (size_t)blockN * Kd, 2u * TILE_BYTES, Kd, N);
  }

  for (int kt = 0; kt < nK; ++kt) {
    const int cur = kt & 1;

    if (kt + 1 < nK) {
      if (wid == 0) {
        const int k1 = (kt + 1) * 32;
        tdm_issue(A + (size_t)blockM * Kd + k1, (unsigned)((cur ^ 1) * TILE_BYTES),
                  Kd, M);
        tdm_issue(B + (size_t)blockN * Kd + k1, (unsigned)(2 * TILE_BYTES + (cur ^ 1) * TILE_BYTES),
                  Kd, N);
      }
      // 2 newer TDM ops may be outstanding; tile `cur` (2 older ops) is done
      __builtin_amdgcn_s_wait_tensorcnt(2);
    } else {
      __builtin_amdgcn_s_wait_tensorcnt(0);
    }
    __syncthreads();

    const unsigned short* as = smem + cur * (128 * LDS_STRIDE);
    const unsigned short* bs = smem + (2 + cur) * (128 * LDS_STRIDE);

    // A 16x32 bf16 frag: lanes 0-15 (M=lr): K 0-7 & 16-23; lanes 16-31: K 8-15 & 24-31
    Frag16 fa[4], fb[2];
    #pragma unroll
    for (int mi = 0; mi < 4; ++mi) {
      const int r = waveRow + mi * 16 + lr;
      fa[mi].q[0] = *(const uint4*)&as[r * LDS_STRIDE + hi * 8];
      fa[mi].q[1] = *(const uint4*)&as[r * LDS_STRIDE + 16 + hi * 8];
    }
    // B 32x16 bf16 frag: lane = column N; lanes 0-15 hold K 0-7 & 16-23, lanes 16-31 K 8-15 & 24-31
    #pragma unroll
    for (int ni = 0; ni < 2; ++ni) {
      const int n = waveCol + ni * 16 + lr;
      fb[ni].q[0] = *(const uint4*)&bs[n * LDS_STRIDE + hi * 8];
      fb[ni].q[1] = *(const uint4*)&bs[n * LDS_STRIDE + 16 + hi * 8];
    }

    #pragma unroll
    for (int mi = 0; mi < 4; ++mi)
      #pragma unroll
      for (int ni = 0; ni < 2; ++ni)
        acc[mi][ni] = __builtin_amdgcn_wmma_f32_16x16x32_bf16(
            false, fa[mi].v, false, fb[ni].v,
            (short)0, acc[mi][ni], false, false);

    __syncthreads();  // all waves done with buf `cur` before TDM overwrites it
  }

  // epilogue: C/D layout: VGPR i -> M=i (lanes 0-15) / M=8+i (lanes 16-31), N=lr
  #pragma unroll
  for (int ni = 0; ni < 2; ++ni) {
    const int col = blockN + waveCol + ni * 16 + lr;
    float bv = 0.0f;
    if (BIAS) bv = bias[col];
    #pragma unroll
    for (int mi = 0; mi < 4; ++mi) {
      const int row0 = blockM + waveRow + mi * 16 + hi * 8;
      #pragma unroll
      for (int i = 0; i < 8; ++i) {
        float v = acc[mi][ni][i] + bv;
        if (RELU) v = v > 0.0f ? v : 0.0f;
        if (WF32)  Cf[(size_t)(row0 + i) * N + col]  = v;
        if (WBF16) Cbf[(size_t)(row0 + i) * N + col] = f2bf(v);
      }
    }
  }
}

// ---- elementwise f32 -> bf16 (n multiple of 1024) ----
__global__ __launch_bounds__(256)
void convert_bf16(const float* __restrict__ in, unsigned short* __restrict__ out) {
  const size_t i = (size_t)blockIdx.x * 256 + threadIdx.x;   // one float4 each
  float4 v = ((const float4*)in)[i];
  uint2 p;
  p.x = pack2bf(v.x, v.y);
  p.y = pack2bf(v.z, v.w);
  ((uint2*)out)[i] = p;
}

// ---- W [K][N] f32 -> Wt [N][K] bf16 (tiled transpose) ----
__global__ __launch_bounds__(256)
void transpose_convert(const float* __restrict__ W, unsigned short* __restrict__ Wt,
                       int K, int N) {
  __shared__ float t[32][33];
  const int n0 = blockIdx.x * 32, k0 = blockIdx.y * 32;
  const int tx = threadIdx.x, ty = threadIdx.y;      // 32 x 8
  #pragma unroll
  for (int i = ty; i < 32; i += 8)
    t[i][tx] = W[(size_t)(k0 + i) * N + n0 + tx];
  __syncthreads();
  #pragma unroll
  for (int i = ty; i < 32; i += 8)
    Wt[(size_t)(n0 + i) * K + k0 + tx] = f2bf(t[tx][i]);
}

// ---- ||c_k||^2 for each center (K=1024, D=256) ----
__global__ __launch_bounds__(256)
void center_norms(const float* __restrict__ centers, float* __restrict__ c2) {
  __shared__ float red[256];
  const int k = blockIdx.x, t = threadIdx.x;
  float v = centers[(size_t)k * 256 + t];
  red[t] = v * v;
  __syncthreads();
  for (int s = 128; s > 0; s >>= 1) {
    if (t < s) red[t] += red[t + s];
    __syncthreads();
  }
  if (t == 0) c2[k] = red[0];
}

// ---- per-row argmin of (c2[k] - 2*S[b][k]) -> one-hot int32 labels ----
__global__ __launch_bounds__(256)
void argmin_onehot(const float* __restrict__ S, const float* __restrict__ c2,
                   int* __restrict__ labels, int K) {
  __shared__ float sd[256];
  __shared__ int   si[256];
  const int b = blockIdx.x, t = threadIdx.x;
  float best = 3.4e38f; int bk = 0;
  for (int k = t; k < K; k += 256) {
    float d = c2[k] - 2.0f * S[(size_t)b * K + k];
    if (d < best || (d == best && k < bk)) { best = d; bk = k; }
  }
  sd[t] = best; si[t] = bk;
  __syncthreads();
  for (int s = 128; s > 0; s >>= 1) {
    if (t < s) {
      float od = sd[t + s]; int ok = si[t + s];
      if (od < sd[t] || (od == sd[t] && ok < si[t])) { sd[t] = od; si[t] = ok; }
    }
    __syncthreads();
  }
  const int win = si[0];
  for (int k = t; k < K; k += 256)
    labels[(size_t)b * K + k] = (k == win) ? 1 : 0;
}

extern "C" void kernel_launch(void* const* d_in, const int* in_sizes, int n_in,
                              void* d_out, int out_size, void* d_ws, size_t ws_size,
                              hipStream_t stream) {
  (void)in_sizes; (void)n_in; (void)out_size; (void)ws_size;

  const float* x   = (const float*)d_in[0];
  const float* We0 = (const float*)d_in[1];  const float* be0 = (const float*)d_in[2];
  const float* We1 = (const float*)d_in[3];  const float* be1 = (const float*)d_in[4];
  const float* We2 = (const float*)d_in[5];  const float* be2 = (const float*)d_in[6];
  const float* We3 = (const float*)d_in[7];  const float* be3 = (const float*)d_in[8];
  const float* Wd0 = (const float*)d_in[9];  const float* bd0 = (const float*)d_in[10];
  const float* Wd1 = (const float*)d_in[11]; const float* bd1 = (const float*)d_in[12];
  const float* Wd2 = (const float*)d_in[13]; const float* bd2 = (const float*)d_in[14];
  const float* Wd3 = (const float*)d_in[15]; const float* bd3 = (const float*)d_in[16];
  const float* centers = (const float*)d_in[17];

  const int M = 8192, K = 1024, DEMB = 256;

  // d_out: reconstruction [8192][1024] f32, embedding [8192][256] f32,
  //        labels [8192][1024] int32
  float* out = (float*)d_out;
  float* rec = out;
  float* emb = out + (size_t)M * 1024;
  int*   lab = (int*)(out + (size_t)M * 1024 + (size_t)M * DEMB);

  // workspace carve-up (256B aligned chunks)
  char* wsp = (char*)d_ws;
  auto alloc = [&](size_t bytes) { char* p = wsp; wsp += (bytes + 255) & ~(size_t)255; return p; };
  unsigned short* bf0   = (unsigned short*)alloc((size_t)M * 2048 * 2);
  unsigned short* bf1   = (unsigned short*)alloc((size_t)M * 2048 * 2);
  unsigned short* xbf   = (unsigned short*)alloc((size_t)M * 1024 * 2);
  unsigned short* embbf = (unsigned short*)alloc((size_t)M * DEMB * 2);
  unsigned short* We0t  = (unsigned short*)alloc((size_t)2048 * 1024 * 2);
  unsigned short* We1t  = (unsigned short*)alloc((size_t)1024 * 2048 * 2);
  unsigned short* We2t  = (unsigned short*)alloc((size_t)512  * 1024 * 2);
  unsigned short* We3t  = (unsigned short*)alloc((size_t)256  * 512  * 2);
  unsigned short* Wd0t  = (unsigned short*)alloc((size_t)512  * 256  * 2);
  unsigned short* Wd1t  = (unsigned short*)alloc((size_t)1024 * 512  * 2);
  unsigned short* Wd2t  = (unsigned short*)alloc((size_t)2048 * 1024 * 2);
  unsigned short* Wd3t  = (unsigned short*)alloc((size_t)1024 * 2048 * 2);
  unsigned short* ctrb  = (unsigned short*)alloc((size_t)K * DEMB * 2);
  float*          Sbuf  = (float*)alloc((size_t)M * K * 4);
  float*          c2b   = (float*)alloc((size_t)K * 4);

  dim3 blk(256);
  dim3 tblk(32, 8);
  const unsigned shmem = 4u * TILE_BYTES;   // 40960 B (double-buffered A+B tiles)
  auto grd  = [](int n, int m) { return dim3(n / 128, m / 128); };
  auto tgrd = [](int Kdim, int Ndim) { return dim3(Ndim / 32, Kdim / 32); };

  // one-time per-call precomputation: bf16 conversions / weight transposes
  convert_bf16<<<dim3((M * 1024) / 1024), blk, 0, stream>>>(x, xbf);
  convert_bf16<<<dim3((K * DEMB) / 1024), blk, 0, stream>>>(centers, ctrb);
  transpose_convert<<<tgrd(1024, 2048), tblk, 0, stream>>>(We0, We0t, 1024, 2048);
  transpose_convert<<<tgrd(2048, 1024), tblk, 0, stream>>>(We1, We1t, 2048, 1024);
  transpose_convert<<<tgrd(1024,  512), tblk, 0, stream>>>(We2, We2t, 1024,  512);
  transpose_convert<<<tgrd( 512,  256), tblk, 0, stream>>>(We3, We3t,  512,  256);
  transpose_convert<<<tgrd( 256,  512), tblk, 0, stream>>>(Wd0, Wd0t,  256,  512);
  transpose_convert<<<tgrd( 512, 1024), tblk, 0, stream>>>(Wd1, Wd1t,  512, 1024);
  transpose_convert<<<tgrd(1024, 2048), tblk, 0, stream>>>(Wd2, Wd2t, 1024, 2048);
  transpose_convert<<<tgrd(2048, 1024), tblk, 0, stream>>>(Wd3, Wd3t, 2048, 1024);
  center_norms<<<dim3(K), blk, 0, stream>>>(centers, c2b);

  // encoder: 1024 -> 2048 -> 1024 -> 512 -> 256
  gemm_tdm<1,1,0,1><<<grd(2048, M), blk, shmem, stream>>>(xbf, We0t, be0, nullptr, bf0, M, 2048, 1024);
  gemm_tdm<1,1,0,1><<<grd(1024, M), blk, shmem, stream>>>(bf0, We1t, be1, nullptr, bf1, M, 1024, 2048);
  gemm_tdm<1,1,0,1><<<grd( 512, M), blk, shmem, stream>>>(bf1, We2t, be2, nullptr, bf0, M,  512, 1024);
  gemm_tdm<0,1,1,1><<<grd( 256, M), blk, shmem, stream>>>(bf0, We3t, be3, emb,  embbf, M,  256,  512);

  // decoder: 256 -> 512 -> 1024 -> 2048 -> 1024
  gemm_tdm<1,1,0,1><<<grd( 512, M), blk, shmem, stream>>>(embbf, Wd0t, bd0, nullptr, bf0, M,  512,  256);
  gemm_tdm<1,1,0,1><<<grd(1024, M), blk, shmem, stream>>>(bf0,   Wd1t, bd1, nullptr, bf1, M, 1024,  512);
  gemm_tdm<1,1,0,1><<<grd(2048, M), blk, shmem, stream>>>(bf1,   Wd2t, bd2, nullptr, bf0, M, 2048, 1024);
  gemm_tdm<0,1,1,0><<<grd(1024, M), blk, shmem, stream>>>(bf0,   Wd3t, bd3, rec,  nullptr, M, 1024, 2048);

  // clustering: S = emb * centers^T, then argmin -> one-hot
  gemm_tdm<0,0,1,0><<<grd(K, M), blk, shmem, stream>>>(embbf, ctrb, nullptr, Sbuf, nullptr, M, K, DEMB);
  argmin_onehot<<<dim3(M), blk, 0, stream>>>(Sbuf, c2b, lab, K);
}
